// TextLSTM_25572235281182
// MI455X (gfx1250) — matrix operand order repaired and checked by reading, not compiled, not run on hardware
//
#include <hip/hip_runtime.h>

// ---------------------------------------------------------------------------
// TextLSTM for MI455X (gfx1250, wave32, WMMA + TDM).
//   1) convert WF/WI/WC/WO and Wout f32 -> bf16 in workspace
//   2) embedding gather X -> xemb bf16 [T][B][E]
//   3) zero h0 (bf16) and c (f32)
//   4) 5x fused LSTM-step: TDM-staged weights in LDS (double-buffered),
//      register-prefetched A, v_wmma_f32_16x16x32_bf16, fused cell update
//   5) classifier GEMM, same TDM/LDS/WMMA structure, bias fused, f32 out
// ---------------------------------------------------------------------------

typedef __attribute__((ext_vector_type(16))) __bf16       v16bf;
typedef __attribute__((ext_vector_type(4)))  __bf16       v4bf;
typedef __attribute__((ext_vector_type(8)))  float        v8f;
typedef __attribute__((ext_vector_type(4)))  unsigned int v4u;
typedef __attribute__((ext_vector_type(8)))  int          v8i;
typedef __attribute__((ext_vector_type(4)))  int          v4i;

#define DEVI __device__ __forceinline__

constexpr int NCLS = 32000;
constexpr int EMBD = 512;
constexpr int HID  = 1024;
constexpr int HE   = 1536;   // HID + EMBD
constexpr int BSZ  = 4096;
constexpr int TLEN = 5;

// K-chunk staged per TDM transfer (2 WMMA K-steps)
constexpr int KCH = 64;
// LDS row: 64 bf16 = 128B data + 16B TDM pad -> 144B (16B aligned, 36 dwords,
// gcd(36,64)=4 -> 16 distinct banks across the 16 columns of a fragment read)
constexpr int ROWB = 144;

#if defined(__has_builtin)
#if __has_builtin(__builtin_amdgcn_tensor_load_to_lds) && \
    __has_builtin(__builtin_amdgcn_s_wait_tensorcnt)
#define USE_TDM 1
#endif
#endif
#ifndef USE_TDM
#define USE_TDM 0
#endif

DEVI float fsigmoid(float x) { return 1.0f / (1.0f + __expf(-x)); }

// --- A fragment: 16x32 bf16 (MxK) from row-major global (contiguous K: lanes 0-15 K=0..15, lanes 16-31 K=16..31).
DEVI v16bf load_a16x32(const __bf16* base, int stride, int kk, int lane) {
  const int row = lane & 15;
  const int off = (lane >> 4) << 4;                 // 0 or 16 elements
  const __bf16* p = base + (size_t)row * stride + kk + off;
  v16bf a;
  ((v4u*)&a)[0] = *(const v4u*)(p);                 // global_load_b128
  ((v4u*)&a)[1] = *(const v4u*)(p + 8);
  return a;
}

// A from the concatenated [h | x] operand; KCH chunks never straddle the seam.
DEVI v16bf load_a_hx(const __bf16* hA, const __bf16* xA, int kk, int lane) {
  if (kk < HID) return load_a16x32(hA, HID, kk, lane);
  return load_a16x32(xA, EMBD, kk - HID, lane);
}

// --- B fragment from LDS-staged tile: rows are columns n (stride ROWB bytes),
// 64 bf16 K-elements per row. lanes 0-15: col=lane, K=ksub+0..15;
// lanes 16-31: col=lane-16, K=ksub+16..31.  -> ds_load_b128 x2
DEVI v16bf lds_b_frag(const char* base, int ksub, int lane) {
  const int col = lane & 15;
  const int off = (lane >> 4) << 4;                 // 0 or 16 elements
  const char* p = base + col * ROWB + (ksub + off) * 2;
  v16bf b;
  ((v4u*)&b)[0] = *(const v4u*)(p);
  ((v4u*)&b)[1] = *(const v4u*)(p + 16);
  return b;
}

DEVI v8f wmma_bf16(v16bf a, v16bf b, v8f c) {
  return __builtin_amdgcn_wmma_f32_16x16x32_bf16(
      false, a, false, b, (short)0, c, false, false);
}

// ---------------------------------------------------------------------------
// TDM descriptor issue (ISA 8.3/8.4): up-to-3D tile, bf16 elements,
// pad_enable=1, pad_interval=4 (32 dwords=128B), pad_amount=3 (4 dwords=16B).
//   d#g1.d0 = data_size(1)<<16 | pad_en<<20 | pad_int(4)<<22 | pad_amt(3)<<25
// This toolchain exposes the 6-argument builtin:
//   (uint32x4 g0, int32x8 g1, int32x4 g2, int32x4 g3, int32x8 g4, i32 cpol)
// ---------------------------------------------------------------------------
DEVI void tdm_load_tile(unsigned lds_byte_addr, unsigned long long gaddr,
                        unsigned tdim0, unsigned tdim1, unsigned tile0,
                        unsigned tile1, unsigned tile2,
                        unsigned long long stride0,
                        unsigned long long stride1, unsigned tdim2) {
#if USE_TDM
  v4u g0;
  g0[0] = 1u;                                    // count=1, user descriptor
  g0[1] = lds_byte_addr;                         // lds_addr
  g0[2] = (unsigned)(gaddr & 0xFFFFFFFFull);     // global_addr[31:0]
  g0[3] = (unsigned)((gaddr >> 32) & 0x01FFFFFFull) | (2u << 30); // type=2
  v8i g1;
  g1[0] = (int)0x07110000u;                      // data_size=2B, pad 16B/128B
  g1[1] = (int)((tdim0 & 0xFFFFu) << 16);        // tensor_dim0 lo16
  g1[2] = (int)((tdim0 >> 16) | ((tdim1 & 0xFFFFu) << 16));
  g1[3] = (int)((tdim1 >> 16) | (tile0 << 16));  // tile_dim0
  g1[4] = (int)(tile1 | (tile2 << 16));          // tile_dim1, tile_dim2
  g1[5] = (int)(unsigned)(stride0 & 0xFFFFFFFFull);
  g1[6] = (int)((unsigned)((stride0 >> 32) & 0xFFFFull) |
                ((unsigned)(stride1 & 0xFFFFull) << 16));
  g1[7] = (int)(unsigned)((stride1 >> 16) & 0xFFFFFFFFull);
  v4i g2;
  g2[0] = (int)tdim2; g2[1] = 0; g2[2] = 0; g2[3] = 0;
  v4i g3;
  g3[0] = 0; g3[1] = 0; g3[2] = 0; g3[3] = 0;
  v8i g4;
  g4[0] = 0; g4[1] = 0; g4[2] = 0; g4[3] = 0;
  g4[4] = 0; g4[5] = 0; g4[6] = 0; g4[7] = 0;
  __builtin_amdgcn_tensor_load_to_lds(g0, g1, g2, g3, g4, 0);
#else
  (void)lds_byte_addr; (void)gaddr; (void)tdim0; (void)tdim1; (void)tile0;
  (void)tile1; (void)tile2; (void)stride0; (void)stride1; (void)tdim2;
#endif
}

DEVI void tdm_wait0() {
#if USE_TDM
  __builtin_amdgcn_s_wait_tensorcnt(0);
#endif
}

// ---------------------------------------------------------------------------
// Utility kernels
// ---------------------------------------------------------------------------

__global__ void k_convert_bf16(const float* __restrict__ src,
                               __bf16* __restrict__ dst, int n4) {
  int i = blockIdx.x * blockDim.x + threadIdx.x;
  if (i >= n4) return;
  float4 f = ((const float4*)src)[i];
  v4bf o;
  o[0] = (__bf16)f.x; o[1] = (__bf16)f.y; o[2] = (__bf16)f.z; o[3] = (__bf16)f.w;
  ((v4bf*)dst)[i] = o;
}

__global__ void k_zero_u32(unsigned int* __restrict__ p, int n) {
  int i = blockIdx.x * blockDim.x + threadIdx.x;
  if (i < n) p[i] = 0u;
}

__global__ void k_embed(const long long* __restrict__ X,   // [B, T] int64
                        const float* __restrict__ Emb,     // [NCLS, EMBD]
                        __bf16* __restrict__ xemb) {       // [T][B][EMBD]
  int i = blockIdx.x * blockDim.x + threadIdx.x;           // over T*B*EMBD/4
  const int e4 = i % (EMBD / 4);
  const int bt = i / (EMBD / 4);
  const int b  = bt % BSZ;
  const int t  = bt / BSZ;
  const long long tok = X[(size_t)b * TLEN + t];
  float4 f = ((const float4*)(Emb + (size_t)tok * EMBD))[e4];
  v4bf o;
  o[0] = (__bf16)f.x; o[1] = (__bf16)f.y; o[2] = (__bf16)f.z; o[3] = (__bf16)f.w;
  ((v4bf*)xemb)[i] = o;
}

// ---------------------------------------------------------------------------
// Fused LSTM step. Block = 128 threads (4 waves), tile M=64 x N=16.
// Weights for all 4 gates staged per K-chunk by ONE 3D TDM transfer
// (64K x 16N x 4 gates, gate stride = HID*HE elements) into double-buffered
// LDS; each wave ds_loads B fragments. A (hx) register-prefetched.
// Grid = (BSZ/64, HID/16).
// ---------------------------------------------------------------------------
constexpr int L_GATEB = 16 * ROWB;       // 2304 B per gate tile
constexpr int L_BUFB  = 4 * L_GATEB;     // 9216 B per buffer

__global__ __launch_bounds__(128) void k_lstm_step(
    const __bf16* __restrict__ hin,   // [B, HID] bf16
    const __bf16* __restrict__ xt,    // [B, EMBD] bf16
    const __bf16* __restrict__ w4,    // [4][HID][HE] bf16 (F,I,C,O)
    const float* __restrict__ bF, const float* __restrict__ bI,
    const float* __restrict__ bC, const float* __restrict__ bO,
    float* __restrict__ c,            // [B, HID] f32, in place
    __bf16* __restrict__ hout)        // [B, HID] bf16
{
  __shared__ __align__(128) char smemB[2 * L_BUFB];

  const int lane   = threadIdx.x & 31;
  const int wave   = threadIdx.x >> 5;
  const int m_base = blockIdx.x * 64 + wave * 16;
  const int n_base = blockIdx.y * 16;

  const __bf16* hA = hin + (size_t)m_base * HID;
  const __bf16* xA = xt  + (size_t)m_base * EMBD;

  const unsigned long long wbase =
      (unsigned long long)(const void*)(w4 + (size_t)n_base * HE);
  // low 32 bits of a shared-aperture flat address == LDS byte offset
  const unsigned smem_lds = (unsigned)(size_t)(const void*)smemB;

  constexpr int NCHUNK = HE / KCH;   // 24

#if USE_TDM
  if (wave == 0)
    tdm_load_tile(smem_lds, wbase, HE, HID, KCH, 16, 4,
                  HE, (unsigned long long)HID * HE, 4);
#endif

  v8f aF = {}, aI = {}, aC = {}, aO = {};
  v16bf a0 = load_a_hx(hA, xA, 0, lane);
  v16bf a1 = load_a_hx(hA, xA, 32, lane);

  for (int it = 0; it < NCHUNK; ++it) {
    const int kk  = it * KCH;
    const int buf = it & 1;
    const int nk  = (it + 1 < NCHUNK) ? kk + KCH : 0;

    // prefetch next A fragments (hidden latency across the chunk's WMMAs)
    v16bf na0 = load_a_hx(hA, xA, nk, lane);
    v16bf na1 = load_a_hx(hA, xA, nk + 32, lane);

#if USE_TDM
    if (wave == 0) tdm_wait0();        // current buffer landed
    __syncthreads();                   // visible to all waves
    if (wave == 0 && it + 1 < NCHUNK)  // stage next chunk into other buffer
      tdm_load_tile(smem_lds + (buf ^ 1) * L_BUFB, wbase + 2ull * nk,
                    HE, HID, KCH, 16, 4, HE, (unsigned long long)HID * HE, 4);
#else
    __syncthreads();                   // previous reads of smem done
    // cooperative fallback copy: 4 gates x 16 rows x 8 x 16B segments
    for (int s = threadIdx.x; s < 512; s += 128) {
      const int g = s >> 7, r = (s >> 3) & 15, q = s & 7;
      const __bf16* src = w4 + (size_t)g * HID * HE +
                          (size_t)(n_base + r) * HE + kk + q * 8;
      *(v4u*)(smemB + g * L_GATEB + r * ROWB + q * 16) = *(const v4u*)src;
    }
    __syncthreads();
#endif

    const char* bb = smemB + (USE_TDM ? buf * L_BUFB : 0);
    aF = wmma_bf16(a0, lds_b_frag(bb + 0 * L_GATEB, 0,  lane), aF);
    aI = wmma_bf16(a0, lds_b_frag(bb + 1 * L_GATEB, 0,  lane), aI);
    aC = wmma_bf16(a0, lds_b_frag(bb + 2 * L_GATEB, 0,  lane), aC);
    aO = wmma_bf16(a0, lds_b_frag(bb + 3 * L_GATEB, 0,  lane), aO);
    aF = wmma_bf16(a1, lds_b_frag(bb + 0 * L_GATEB, 32, lane), aF);
    aI = wmma_bf16(a1, lds_b_frag(bb + 1 * L_GATEB, 32, lane), aI);
    aC = wmma_bf16(a1, lds_b_frag(bb + 2 * L_GATEB, 32, lane), aC);
    aO = wmma_bf16(a1, lds_b_frag(bb + 3 * L_GATEB, 32, lane), aO);

    a0 = na0; a1 = na1;
  }

  // Epilogue: C/D layout -> n = lane&15, rows m = r + 8*(lane>>4)
  const int   n   = n_base + (lane & 15);
  const int   m0  = m_base + ((lane >> 4) << 3);
  const float vbF = bF[n], vbI = bI[n], vbC = bC[n], vbO = bO[n];
#pragma unroll
  for (int r = 0; r < 8; ++r) {
    const size_t idx = (size_t)(m0 + r) * HID + n;
    const float f  = fsigmoid(aF[r] + vbF);
    const float i  = fsigmoid(aI[r] + vbI);
    const float g  = tanhf(aC[r] + vbC);
    const float o  = fsigmoid(aO[r] + vbO);
    const float cn = f * c[idx] + i * g;
    c[idx]    = cn;
    hout[idx] = (__bf16)(o * tanhf(cn));
  }
}

// ---------------------------------------------------------------------------
// Classifier GEMM: out = h @ Wout.T + bout. Block = 256 threads
// (8 waves: 4 along M x 2 along N), tile 64x32. Wout tile (64K x 32N) staged
// per chunk by one 2D TDM transfer, double-buffered. Grid = (64, 1000).
// ---------------------------------------------------------------------------
constexpr int O_BUFB = 32 * ROWB;        // 4608 B per buffer

__global__ __launch_bounds__(256) void k_out_gemm(
    const __bf16* __restrict__ h,      // [B, HID] bf16
    const __bf16* __restrict__ wout,   // [NCLS, HID] bf16
    const float* __restrict__ bout,    // [NCLS]
    float* __restrict__ out)           // [B, NCLS] f32
{
  __shared__ __align__(128) char smemB[2 * O_BUFB];

  const int lane   = threadIdx.x & 31;
  const int wave   = threadIdx.x >> 5;
  const int wm     = wave & 3;
  const int wn     = wave >> 2;
  const int m_base = blockIdx.x * 64 + wm * 16;
  const int nb0    = blockIdx.y * 32;            // block's 32-column panel

  const __bf16* hA = h + (size_t)m_base * HID;
  const unsigned long long wbase =
      (unsigned long long)(const void*)(wout + (size_t)nb0 * HID);
  const unsigned smem_lds = (unsigned)(size_t)(const void*)smemB;

  constexpr int NCHUNK = HID / KCH;   // 16

#if USE_TDM
  if (wave == 0)
    tdm_load_tile(smem_lds, wbase, HID, NCLS, KCH, 32, 0, HID, 0, 0);
#endif

  v8f acc = {};
  v16bf a0 = load_a16x32(hA, HID, 0, lane);
  v16bf a1 = load_a16x32(hA, HID, 32, lane);

  for (int it = 0; it < NCHUNK; ++it) {
    const int kk  = it * KCH;
    const int buf = it & 1;
    const int nk  = (it + 1 < NCHUNK) ? kk + KCH : 0;

    v16bf na0 = load_a16x32(hA, HID, nk, lane);
    v16bf na1 = load_a16x32(hA, HID, nk + 32, lane);

#if USE_TDM
    if (wave == 0) tdm_wait0();
    __syncthreads();
    if (wave == 0 && it + 1 < NCHUNK)
      tdm_load_tile(smem_lds + (buf ^ 1) * O_BUFB, wbase + 2ull * nk,
                    HID, NCLS, KCH, 32, 0, HID, 0, 0);
#else
    __syncthreads();
    for (int s = threadIdx.x; s < 256; s += 256) {
      const int r = s >> 3, q = s & 7;             // 32 rows x 8 segments
      const __bf16* src = wout + (size_t)(nb0 + r) * HID + kk + q * 8;
      *(v4u*)(smemB + r * ROWB + q * 16) = *(const v4u*)src;
    }
    __syncthreads();
#endif

    const char* bb = smemB + (USE_TDM ? buf * O_BUFB : 0) + wn * 16 * ROWB;
    acc = wmma_bf16(a0, lds_b_frag(bb, 0,  lane), acc);
    acc = wmma_bf16(a1, lds_b_frag(bb, 32, lane), acc);

    a0 = na0; a1 = na1;
  }

  const int   n    = nb0 + wn * 16 + (lane & 15);
  const int   m0   = m_base + ((lane >> 4) << 3);
  const float bias = bout[n];
#pragma unroll
  for (int r = 0; r < 8; ++r)
    out[(size_t)(m0 + r) * NCLS + n] = acc[r] + bias;
}

// ---------------------------------------------------------------------------
// Launch
// ---------------------------------------------------------------------------
extern "C" void kernel_launch(void* const* d_in, const int* in_sizes, int n_in,
                              void* d_out, int out_size, void* d_ws, size_t ws_size,
                              hipStream_t stream) {
  (void)in_sizes; (void)n_in; (void)out_size; (void)ws_size;

  const long long* X    = (const long long*)d_in[0];
  const float*     Emb  = (const float*)d_in[1];
  const float*     WF   = (const float*)d_in[2];
  const float*     bF   = (const float*)d_in[3];
  const float*     WI   = (const float*)d_in[4];
  const float*     bI   = (const float*)d_in[5];
  const float*     WC   = (const float*)d_in[6];
  const float*     bC   = (const float*)d_in[7];
  const float*     WO   = (const float*)d_in[8];
  const float*     bO   = (const float*)d_in[9];
  const float*     Wout = (const float*)d_in[10];
  const float*     bout = (const float*)d_in[11];
  float* out = (float*)d_out;

  char*  ws  = (char*)d_ws;
  size_t off = 0;
  auto carve = [&](size_t bytes) -> char* {
    char* p = ws + off;
    off = (off + bytes + 255) & ~(size_t)255;
    return p;
  };
  __bf16* w4    = (__bf16*)carve((size_t)4 * HID * HE * 2);
  __bf16* woutb = (__bf16*)carve((size_t)NCLS * HID * 2);
  __bf16* xemb  = (__bf16*)carve((size_t)TLEN * BSZ * EMBD * 2);
  __bf16* h0    = (__bf16*)carve((size_t)BSZ * HID * 2);
  __bf16* h1    = (__bf16*)carve((size_t)BSZ * HID * 2);
  float*  cbuf  = (float*)carve((size_t)BSZ * HID * 4);

  const int TPB = 256;

  {
    const int n4 = HID * HE / 4;
    k_convert_bf16<<<(n4 + TPB - 1) / TPB, TPB, 0, stream>>>(WF, w4 + (size_t)0 * HID * HE, n4);
    k_convert_bf16<<<(n4 + TPB - 1) / TPB, TPB, 0, stream>>>(WI, w4 + (size_t)1 * HID * HE, n4);
    k_convert_bf16<<<(n4 + TPB - 1) / TPB, TPB, 0, stream>>>(WC, w4 + (size_t)2 * HID * HE, n4);
    k_convert_bf16<<<(n4 + TPB - 1) / TPB, TPB, 0, stream>>>(WO, w4 + (size_t)3 * HID * HE, n4);
    const int n4o = NCLS * HID / 4;
    k_convert_bf16<<<(n4o + TPB - 1) / TPB, TPB, 0, stream>>>(Wout, woutb, n4o);
  }
  {
    const int n4 = TLEN * BSZ * EMBD / 4;
    k_embed<<<(n4 + TPB - 1) / TPB, TPB, 0, stream>>>(X, Emb, xemb);
  }
  {
    const int nh = BSZ * HID / 2;
    const int nc = BSZ * HID;
    k_zero_u32<<<(nh + TPB - 1) / TPB, TPB, 0, stream>>>((unsigned int*)h0, nh);
    k_zero_u32<<<(nc + TPB - 1) / TPB, TPB, 0, stream>>>((unsigned int*)cbuf, nc);
  }
  {
    dim3 grid(BSZ / 64, HID / 16);
    __bf16* hin = h0; __bf16* hnx = h1;
    for (int t = 0; t < TLEN; ++t) {
      k_lstm_step<<<grid, 128, 0, stream>>>(
          hin, xemb + (size_t)t * BSZ * EMBD, w4, bF, bI, bC, bO, cbuf, hnx);
      __bf16* tmp = hin; hin = hnx; hnx = tmp;
    }
    dim3 ogrid(BSZ / 64, NCLS / 32);
    k_out_gemm<<<ogrid, 256, 0, stream>>>(hin, woutb, bout, out);
  }
}
